// SelfAttention_34273839022954
// MI455X (gfx1250) — compile-verified, matrix operand order checked by reading, not run on hardware
//
#include <hip/hip_runtime.h>
#include <hip/hip_bf16.h>

// ---------------- problem constants ----------------
#define BATCH 4
#define SEQ   1024
#define HID   1024
#define NHEAD 16
#define DHEAD 64
#define BS    (BATCH * SEQ)          // 4096 rows

// ---------------- vector types for WMMA ----------------
typedef __attribute__((ext_vector_type(16))) __bf16 v16bf;
typedef __attribute__((ext_vector_type(8)))  __bf16 v8bf;
typedef __attribute__((ext_vector_type(8)))  float  v8f;

union V16 { v16bf v; v8bf h[2]; };

struct BiasPack { const float* p[6]; };

// ---------------- fp32 -> bf16 convert ----------------
__global__ __launch_bounds__(256) void cvt_f32_bf16(const float* __restrict__ src,
                                                    __bf16* __restrict__ dst, int n) {
    int i = blockIdx.x * blockDim.x + threadIdx.x;
    if (i < n) dst[i] = (__bf16)src[i];
}

// ---------------- projection: Y = x @ W^T + b ----------------
// One wave computes a 16(M) x 64(N) tile. blockDim = 256 (8 waves).
// grid = (HID/64, BS/128, 6)  -> blockIdx.z selects which of the 6 projections.
// Outputs (bf16) in workspace, region widx of size BS*HID:
//   widx 0,1,3,4 (Q1,K1,Q2,K2): [B, NH, S, DH]
//   widx 2,5      (V1,V2)      : [B, NH, DH, S]   (transposed for P@V)
__global__ __launch_bounds__(256) void proj_kernel(
    const __bf16* __restrict__ xb,      // [BS, HID] bf16
    const __bf16* __restrict__ wb,      // [6, HID, HID] bf16
    BiasPack bp,
    __bf16* __restrict__ outq)          // [6, BS*HID] region base
{
    const int widx = blockIdx.z;
    const int lane = threadIdx.x & 31;
    const int wid  = threadIdx.x >> 5;
    const int lr   = lane & 15;          // row-in-tile for A, col-in-tile for B/C/D
    const int kb   = (lane >> 4) * 8;    // K chunk base per 16-bit A/B layout

    const int m0 = (blockIdx.y * 8 + wid) * 16;
    const int n0 = blockIdx.x * 64;

    const __bf16* W    = wb + (size_t)widx * HID * HID;
    const float*  bias = bp.p[widx];

    const __bf16* arow = xb + (size_t)(m0 + lr) * HID + kb;
    const __bf16* wrow = W  + (size_t)(n0 + lr) * HID + kb;

    v8f acc[4] = {v8f{}, v8f{}, v8f{}, v8f{}};

    for (int kk = 0; kk < HID; kk += 32) {
        V16 a;
        a.h[0] = *(const v8bf*)(arow + kk);
        a.h[1] = *(const v8bf*)(arow + kk + 16);
        V16 bt[4];
        #pragma unroll
        for (int nt = 0; nt < 4; ++nt) {
            const __bf16* p = wrow + (size_t)nt * 16 * HID + kk;
            bt[nt].h[0] = *(const v8bf*)(p);
            bt[nt].h[1] = *(const v8bf*)(p + 16);
        }
        #pragma unroll
        for (int nt = 0; nt < 4; ++nt) {
            acc[nt] = __builtin_amdgcn_wmma_f32_16x16x32_bf16(
                false, a.v, false, bt[nt].v, (short)0, acc[nt], false, false);
        }
    }

    // epilogue: add bias, convert, scatter into attention-friendly layout
    const bool isV = (widx == 2) || (widx == 5);
    __bf16* dst = outq + (size_t)widx * BS * HID;
    #pragma unroll
    for (int r = 0; r < 8; ++r) {
        const int m = m0 + r + kb;            // kb also == row offset (0 or 8)
        const int b = m >> 10;                // / SEQ
        const int s = m & (SEQ - 1);
        #pragma unroll
        for (int nt = 0; nt < 4; ++nt) {
            const int n = n0 + nt * 16 + lr;
            const int h = n >> 6;             // / DHEAD
            const int d = n & (DHEAD - 1);
            float v = acc[nt][r] + bias[n];
            size_t idx;
            if (isV) idx = ((size_t)(b * NHEAD + h) * DHEAD + d) * SEQ + s;
            else     idx = ((size_t)(b * NHEAD + h) * SEQ + s) * DHEAD + d;
            dst[idx] = (__bf16)v;
        }
    }
}

// ---------------- flash attention ----------------
// One wave handles 16 queries of one (b,h). grid = (SEQ/16, B*NH), block = 32.
// Q,K: [B*NH, S, DH] bf16 ; Vt: [B*NH, DH, S] bf16 ; out: [B, S, HID] fp32.
__global__ __launch_bounds__(32) void attn_kernel(
    const __bf16* __restrict__ Q,
    const __bf16* __restrict__ K,
    const __bf16* __restrict__ Vt,
    float* __restrict__ out)
{
    __shared__ __bf16 Pl[16 * 40];    // 16 rows x 32 cols, stride 40 (16B-aligned chunks)

    const int lane = threadIdx.x;
    const int lr   = lane & 15;
    const int kb   = (lane >> 4) * 8;

    const int bh = blockIdx.y;
    const int q0 = blockIdx.x * 16;

    const __bf16* Qb = Q  + (size_t)bh * SEQ * DHEAD;
    const __bf16* Kb = K  + (size_t)bh * SEQ * DHEAD;
    const __bf16* Vb = Vt + (size_t)bh * DHEAD * SEQ;

    // Q tile as two A operands (DH = 64 = 2 x K32), resident whole loop
    V16 aq0, aq1;
    {
        const __bf16* qrow = Qb + (size_t)(q0 + lr) * DHEAD + kb;
        aq0.h[0] = *(const v8bf*)(qrow);
        aq0.h[1] = *(const v8bf*)(qrow + 16);
        aq1.h[0] = *(const v8bf*)(qrow + 32);
        aq1.h[1] = *(const v8bf*)(qrow + 48);
    }

    float mrow[8], lrow[8];
    v8f o[4] = {v8f{}, v8f{}, v8f{}, v8f{}};
    #pragma unroll
    for (int r = 0; r < 8; ++r) { mrow[r] = -1e30f; lrow[r] = 0.0f; }

    const float scale = 0.125f;   // 1/sqrt(64)

    for (int kt = 0; kt < SEQ; kt += 32) {
        // ---- scores: two 16x16 tiles (32 keys) ----
        v8f s[2];
        #pragma unroll
        for (int sub = 0; sub < 2; ++sub) {
            const __bf16* krow = Kb + (size_t)(kt + sub * 16 + lr) * DHEAD + kb;
            V16 bk0, bk1;
            bk0.h[0] = *(const v8bf*)(krow);
            bk0.h[1] = *(const v8bf*)(krow + 16);
            bk1.h[0] = *(const v8bf*)(krow + 32);
            bk1.h[1] = *(const v8bf*)(krow + 48);
            v8f z = {};
            z = __builtin_amdgcn_wmma_f32_16x16x32_bf16(false, aq0.v, false, bk0.v,
                                                        (short)0, z, false, false);
            z = __builtin_amdgcn_wmma_f32_16x16x32_bf16(false, aq1.v, false, bk1.v,
                                                        (short)0, z, false, false);
            s[sub] = z;
        }

        // ---- online softmax (row stats live per half-wave lane group) ----
        #pragma unroll
        for (int r = 0; r < 8; ++r) {
            float s0 = s[0][r] * scale;
            float s1 = s[1][r] * scale;
            float t = fmaxf(s0, s1);
            #pragma unroll
            for (int off = 1; off < 16; off <<= 1)
                t = fmaxf(t, __shfl_xor(t, off, 32));
            const float mn    = fmaxf(mrow[r], t);
            const float alpha = __expf(mrow[r] - mn);
            mrow[r] = mn;
            const float p0 = __expf(s0 - mn);
            const float p1 = __expf(s1 - mn);
            float rs = p0 + p1;
            #pragma unroll
            for (int off = 1; off < 16; off <<= 1)
                rs += __shfl_xor(rs, off, 32);
            lrow[r] = lrow[r] * alpha + rs;
            #pragma unroll
            for (int j = 0; j < 4; ++j) o[j][r] *= alpha;

            const int row = r + kb;   // kb == 0 or 8: C-layout row offset
            Pl[row * 40 + lr]      = (__bf16)p0;
            Pl[row * 40 + 16 + lr] = (__bf16)p1;
        }
        __syncthreads();

        // ---- reload P in A layout, accumulate P @ V ----
        V16 pa;
        pa.h[0] = *(const v8bf*)(&Pl[lr * 40 + kb]);
        pa.h[1] = *(const v8bf*)(&Pl[lr * 40 + kb + 16]);
        #pragma unroll
        for (int j = 0; j < 4; ++j) {
            const __bf16* vrow = Vb + (size_t)(j * 16 + lr) * SEQ + kt + kb;
            V16 bv;
            bv.h[0] = *(const v8bf*)(vrow);
            bv.h[1] = *(const v8bf*)(vrow + 16);
            o[j] = __builtin_amdgcn_wmma_f32_16x16x32_bf16(false, pa.v, false, bv.v,
                                                           (short)0, o[j], false, false);
        }
        __syncthreads();
    }

    // ---- epilogue: normalize and store fp32 ----
    const int b = bh >> 4;          // / NHEAD
    const int h = bh & (NHEAD - 1);
    #pragma unroll
    for (int r = 0; r < 8; ++r) {
        const int q   = q0 + r + kb;
        const float inv = 1.0f / lrow[r];
        #pragma unroll
        for (int j = 0; j < 4; ++j) {
            const int d = j * 16 + lr;
            out[((size_t)(b * SEQ + q)) * HID + h * DHEAD + d] = o[j][r] * inv;
        }
    }
}

// ---------------- host launcher ----------------
extern "C" void kernel_launch(void* const* d_in, const int* in_sizes, int n_in,
                              void* d_out, int out_size, void* d_ws, size_t ws_size,
                              hipStream_t stream) {
    (void)in_sizes; (void)n_in; (void)out_size; (void)ws_size;

    // input order: x, (Wq,bq), (Wk,bk), (Wv,bv), (Wq2,bq2), (Wk2,bk2), (Wv2,bv2)
    const float* x = (const float*)d_in[0];
    const float* W[6];
    BiasPack bp;
    for (int i = 0; i < 6; ++i) {
        W[i]    = (const float*)d_in[1 + 2 * i];
        bp.p[i] = (const float*)d_in[2 + 2 * i];
    }

    // workspace layout (bf16 regions, all 16B aligned):
    //   [0, 8MB)        x_bf16         [BS, HID]
    //   [8MB, 20MB)     W_bf16 x6      [6, HID, HID]
    //   [20MB, 68MB)    Q1,K1,V1t,Q2,K2,V2t  (each BS*HID bf16)
    char* ws = (char*)d_ws;
    const size_t XB_BYTES = (size_t)BS * HID * sizeof(__bf16);        // 8 MB
    const size_t WB_BYTES = (size_t)6 * HID * HID * sizeof(__bf16);   // 12 MB
    __bf16* xb  = (__bf16*)ws;
    __bf16* wb  = (__bf16*)(ws + XB_BYTES);
    __bf16* qkv = (__bf16*)(ws + XB_BYTES + WB_BYTES);

    // 1) convert fp32 -> bf16
    {
        const int nx = BS * HID;
        cvt_f32_bf16<<<(nx + 255) / 256, 256, 0, stream>>>(x, xb, nx);
        const int nw = HID * HID;
        for (int i = 0; i < 6; ++i)
            cvt_f32_bf16<<<(nw + 255) / 256, 256, 0, stream>>>(W[i], wb + (size_t)i * nw, nw);
    }

    // 2) six projections via WMMA
    proj_kernel<<<dim3(HID / 64, BS / 128, 6), 256, 0, stream>>>(xb, wb, bp, qkv);

    // 3) flash attention: out5 = attn(Q1, K2, V2t), out3 = attn(Q2, K1, V1t)
    const size_t R = (size_t)BS * HID;
    float* out = (float*)d_out;
    attn_kernel<<<dim3(SEQ / 16, BATCH * NHEAD), 32, 0, stream>>>(
        qkv + 0 * R, qkv + 4 * R, qkv + 5 * R, out);
    attn_kernel<<<dim3(SEQ / 16, BATCH * NHEAD), 32, 0, stream>>>(
        qkv + 3 * R, qkv + 1 * R, qkv + 2 * R, out + R);
}
